// Mask2FormerDecoder_71305047048434
// MI455X (gfx1250) — matrix operand order repaired and checked by reading, not compiled, not run on hardware
//
// MI455X (gfx1250) deformable-DETR decoder. All GEMMs on v_wmma_f32_16x16x32_f16
// (wave32, f32 accum), 2x2 register tiling (32x32 per wave), guard-free inner loop.
#include <hip/hip_runtime.h>
#include <cstring>
#include <cstdint>

#define D_    256
#define NH_   8
#define DH_   32
#define DFF_  2048
#define NL_   6
#define LL_   3
#define PP_   4
#define BB_   8
#define NQ_   300
#define LEN_  13125
#define MQ    (BB_ * NQ_)          // 2400 query rows
#define MKV   (BB_ * LEN_)         // 105000 key rows
#define KPAD  320                  // padded K for softmax@V (multiple of 32)

typedef __attribute__((ext_vector_type(16))) _Float16 v16h;
typedef __attribute__((ext_vector_type(8)))  _Float16 v8h;
typedef __attribute__((ext_vector_type(8)))  float    v8f;
typedef _Float16 h16;

// ---------------------------------------------------------------- GEMM params
struct GemmParams {
  const h16* A;  long long lda,  aZb,  aZh;     // A: M x K, f16 row-major
  const h16* B;  long long ldb,  bZb,  bZh;     // B: weight layout N x K (x @ W^T)
  float*     C;  long long ldc,  cZb,  cZh;     // optional f32 out [m][n]
  h16*       C16;long long ldc16,c16Zb,c16Zh;   // optional f16 out [m][n]
  h16*       CT; long long ldct, ctZb, ctZh;    // optional f16 transposed out [n][m]
  const float* bias;                            // optional, indexed by n
  const float* res; long long ldres, resZb, resZh; // optional residual (f32)
  const unsigned char* mask;                    // optional per-row m: *(1-mask[m])
  float alpha;
  int M, N, K, zInner, relu;
};

// D = A x B(+C). Each wave computes a 32x32 tile (2x2 WMMA accumulators);
// block = 4 waves covering 128 N-columns.
// WMMA operand layouts (wave32) per ISA 7.12.2:
//  A f16 16x32: lane L holds row m=L&15; two contiguous 8-half runs at
//               k = kb+8*(L>>4) and k = kb+16+8*(L>>4)
//  B f16 32x16: lane L holds col n=L&15; two contiguous 8-half runs
//               W[n][kb+8*(L>>4)] and W[n][kb+16+8*(L>>4)]  (W stored N x K)
//  C f32: VGPR r -> row r + 8*(L>>4), col L&15.
__global__ __launch_bounds__(128) void k_gemm(GemmParams p) {
  const int lane = threadIdx.x & 31;
  const int wave = threadIdx.x >> 5;
  const int ntb  = (blockIdx.y * 4 + wave) * 32;     // wave N base
  if (ntb >= p.N) return;                            // wave-uniform
  const int mtb  = blockIdx.x * 32;
  const int zb = blockIdx.z / p.zInner;
  const int zh = blockIdx.z % p.zInner;
  const h16* A = p.A + zb * p.aZb + zh * p.aZh;
  const h16* B = p.B + zb * p.bZb + zh * p.bZh;

  const int hf  = lane >> 4;
  const int l15 = lane & 15;

  const h16* arow[2]; const h16* brow[2]; int ncol[2];
  #pragma unroll
  for (int t = 0; t < 2; ++t) {
    int mr = mtb + t * 16 + l15;
    int mc = (mr < p.M) ? mr : (p.M - 1);
    arow[t] = A + (long long)mc * p.lda + hf * 8;
    int nc = ntb + t * 16 + l15;  ncol[t] = nc;
    int ncc = (nc < p.N) ? nc : (p.N - 1);
    brow[t] = B + (long long)ncc * p.ldb + hf * 8;
  }

  v8f acc[2][2] = {};
  #pragma unroll 2
  for (int kb = 0; kb < p.K; kb += 32) {
    v16h a[2], b[2];
    #pragma unroll
    for (int t = 0; t < 2; ++t) {
      v8h alo = *(const v8h*)(arow[t] + kb);
      v8h ahi = *(const v8h*)(arow[t] + kb + 16);
      a[t] = __builtin_shufflevector(alo, ahi, 0,1,2,3,4,5,6,7,8,9,10,11,12,13,14,15);
      v8h blo = *(const v8h*)(brow[t] + kb);
      v8h bhi = *(const v8h*)(brow[t] + kb + 16);
      b[t] = __builtin_shufflevector(blo, bhi, 0,1,2,3,4,5,6,7,8,9,10,11,12,13,14,15);
    }
    #pragma unroll
    for (int mi = 0; mi < 2; ++mi)
      #pragma unroll
      for (int ni = 0; ni < 2; ++ni)
        acc[mi][ni] = __builtin_amdgcn_wmma_f32_16x16x32_f16(
            false, a[mi], false, b[ni], (short)0, acc[mi][ni], false, false);
  }

  // epilogue
  #pragma unroll
  for (int ni = 0; ni < 2; ++ni) {
    int nc = ncol[ni];
    if (nc >= p.N) continue;
    const float bv = p.bias ? p.bias[nc] : 0.f;
    #pragma unroll
    for (int mi = 0; mi < 2; ++mi) {
      const int mbase = mtb + mi * 16 + 8 * hf;      // rows mbase..mbase+7
      float vals[8];
      #pragma unroll
      for (int r = 0; r < 8; ++r) {
        int m = mbase + r;
        float t = acc[mi][ni][r] * p.alpha + bv;
        if (p.relu) t = fmaxf(t, 0.f);
        if (p.mask && m < p.M) t *= (1.f - (float)p.mask[m]);
        if (p.res && m < p.M)
          t += p.res[zb * p.resZb + zh * p.resZh + (long long)m * p.ldres + nc];
        vals[r] = t;
      }
      if (p.C) {
        #pragma unroll
        for (int r = 0; r < 8; ++r) { int m = mbase + r;
          if (m < p.M) p.C[zb*p.cZb + zh*p.cZh + (long long)m*p.ldc + nc] = vals[r]; }
      }
      if (p.C16) {
        #pragma unroll
        for (int r = 0; r < 8; ++r) { int m = mbase + r;
          if (m < p.M) p.C16[zb*p.c16Zb + zh*p.c16Zh + (long long)m*p.ldc16 + nc] = (h16)vals[r]; }
      }
      if (p.CT) {
        h16* ct = p.CT + zb*p.ctZb + zh*p.ctZh + (long long)nc * p.ldct + mbase;
        if (mbase + 8 <= p.M) {
          v8h pk;
          #pragma unroll
          for (int r = 0; r < 8; ++r) pk[r] = (h16)vals[r];
          *(v8h*)ct = pk;                             // 16B-aligned contiguous store
        } else {
          #pragma unroll
          for (int r = 0; r < 8; ++r) if (mbase + r < p.M) ct[r] = (h16)vals[r];
        }
      }
    }
  }
}

// ---------------------------------------------------------------- elementwise
__global__ void k_f32_to_f16(const float* __restrict__ x, h16* __restrict__ y,
                             long long n) {
  long long i = (long long)blockIdx.x * blockDim.x + threadIdx.x;
  long long s = (long long)gridDim.x * blockDim.x;
  for (; i < n; i += s) y[i] = (h16)x[i];
}

__global__ void k_zero16(h16* __restrict__ y, long long n) {
  long long i = (long long)blockIdx.x * blockDim.x + threadIdx.x;
  long long s = (long long)gridDim.x * blockDim.x;
  for (; i < n; i += s) y[i] = (h16)0.f;
}

// LayerNorm over D_=256; optionally f32 out, f16 out, and f16(y + addsrc).
__global__ __launch_bounds__(D_) void k_layernorm(const float* __restrict__ x,
    const float* __restrict__ g, const float* __restrict__ be,
    float* y32, h16* y16, const float* addsrc, h16* add16) {
  int row = blockIdx.x, t = threadIdx.x;
  __shared__ float red[D_];
  long long idx = (long long)row * D_ + t;
  float v = x[idx];
  red[t] = v; __syncthreads();
  for (int s = D_/2; s > 0; s >>= 1) { if (t < s) red[t] += red[t+s]; __syncthreads(); }
  float mu = red[0] / (float)D_; __syncthreads();
  float d = v - mu;
  red[t] = d * d; __syncthreads();
  for (int s = D_/2; s > 0; s >>= 1) { if (t < s) red[t] += red[t+s]; __syncthreads(); }
  float var = red[0] / (float)D_;
  float y = d * rsqrtf(var + 1e-5f) * g[t] + be[t];
  if (y32)   y32[idx] = y;
  if (y16)   y16[idx] = (h16)y;
  if (add16) add16[idx] = (h16)(y + addsrc[idx]);
}

// MHA softmax: rows of 300 (f32, stride 304) -> f16 rows padded to 320 with zeros.
__global__ __launch_bounds__(128) void k_softmax_mha(const float* __restrict__ S,
                                                     h16* __restrict__ A) {
  int row = blockIdx.x, t = threadIdx.x;
  const float* s = S + (long long)row * 304;
  __shared__ float red[128];
  float mx = -3.4e38f;
  for (int j = t; j < NQ_; j += 128) mx = fmaxf(mx, s[j]);
  red[t] = mx; __syncthreads();
  for (int st = 64; st > 0; st >>= 1) { if (t < st) red[t] = fmaxf(red[t], red[t+st]); __syncthreads(); }
  mx = red[0]; __syncthreads();
  float sum = 0.f;
  for (int j = t; j < NQ_; j += 128) sum += __expf(s[j] - mx);
  red[t] = sum; __syncthreads();
  for (int st = 64; st > 0; st >>= 1) { if (t < st) red[t] += red[t+st]; __syncthreads(); }
  float inv = 1.f / red[0];
  h16* a = A + (long long)row * KPAD;
  for (int j = t; j < KPAD; j += 128)
    a[j] = (j < NQ_) ? (h16)(__expf(s[j] - mx) * inv) : (h16)0.f;
}

// softmax over the 12 (L*P) attention weights per (m, head)
__global__ void k_attw_softmax(const float* __restrict__ raw, float* __restrict__ aw) {
  int idx = blockIdx.x * blockDim.x + threadIdx.x;
  if (idx >= MQ * NH_) return;
  int m = idx / NH_, h = idx % NH_;
  const float* r = raw + (long long)m * (NH_*LL_*PP_) + h * (LL_*PP_);
  float mx = r[0];
  for (int j = 1; j < 12; ++j) mx = fmaxf(mx, r[j]);
  float e[12], s = 0.f;
  for (int j = 0; j < 12; ++j) { e[j] = __expf(r[j] - mx); s += e[j]; }
  float inv = 1.f / s;
  float* o = aw + (long long)m * 96 + h * 12;
  for (int j = 0; j < 12; ++j) o[j] = e[j] * inv;
}

// multi-scale deformable bilinear sampling (static shapes 100/50/25).
__global__ __launch_bounds__(256) void k_msdeform_sample(const h16* __restrict__ v,
    const float* __restrict__ off, const float* __restrict__ aw,
    const float* __restrict__ qrp, h16* __restrict__ samp) {
  int m  = blockIdx.x;               // b*300+q
  int t  = threadIdx.x;              // h*32+d
  int hh = t >> 5, d = t & 31;
  int b  = m / NQ_;
  float rx = qrp[m*2 + 0], ry = qrp[m*2 + 1];
  const int Hs[3] = {100, 50, 25}, Ws[3] = {100, 50, 25}, st[3] = {0, 10000, 12500};
  float acc = 0.f;
  for (int l = 0; l < 3; ++l) {
    int Hl = Hs[l], Wl = Ws[l];
    const h16* vb = v + ((long long)(b * LEN_ + st[l]) * D_) + hh * DH_ + d;
    for (int p = 0; p < PP_; ++p) {
      long long ob = (long long)m * 192 + ((hh * LL_ + l) * PP_ + p) * 2;
      float lx = rx + off[ob + 0] / (float)Wl;
      float ly = ry + off[ob + 1] / (float)Hl;
      float w  = aw[(long long)m * 96 + hh * 12 + l * PP_ + p];
      float x = lx * Wl - 0.5f, y = ly * Hl - 0.5f;
      float x0f = floorf(x), y0f = floorf(y);
      int x0 = (int)x0f, y0 = (int)y0f;
      float wx1 = x - x0f, wy1 = y - y0f;
      auto g = [&](int xi, int yi) -> float {
        bool valid = (xi >= 0) && (xi < Wl) && (yi >= 0) && (yi < Hl);
        int xc = xi < 0 ? 0 : (xi > Wl-1 ? Wl-1 : xi);
        int yc = yi < 0 ? 0 : (yi > Hl-1 ? Hl-1 : yi);
        float val = (float)vb[(long long)(yc * Wl + xc) * D_];
        return valid ? val : 0.f;
      };
      float sval = g(x0,   y0  ) * (1.f-wx1) * (1.f-wy1)
                 + g(x0+1, y0  ) * wx1       * (1.f-wy1)
                 + g(x0,   y0+1) * (1.f-wx1) * wy1
                 + g(x0+1, y0+1) * wx1       * wy1;
      acc += sval * w;
    }
  }
  samp[(long long)m * D_ + hh * DH_ + d] = (h16)acc;
}

// ref head stage 2: sigmoid(r1 @ w2^T + b2); writes qrp and (optionally) output
// section 3 (L-repeated).
__global__ void k_refhead2(const h16* __restrict__ r1, const float* __restrict__ w2,
                           const float* __restrict__ b2, float* __restrict__ qrp,
                           float* out3) {
  int m = blockIdx.x * blockDim.x + threadIdx.x;
  if (m >= MQ) return;
  for (int j = 0; j < 2; ++j) {
    float s = b2[j];
    const h16* rr = r1 + (long long)m * D_;
    const float* ww = w2 + j * D_;
    for (int k = 0; k < D_; ++k) s += (float)rr[k] * ww[k];
    float sg = 1.f / (1.f + __expf(-s));
    qrp[m*2 + j] = sg;
    if (out3) for (int l = 0; l < LL_; ++l) out3[(long long)m*6 + l*2 + j] = sg;
  }
}

// ---------------------------------------------------------------- host side
static inline GemmParams G0() {
  GemmParams p; memset(&p, 0, sizeof(p));
  p.alpha = 1.f; p.zInner = 1;
  return p;
}
static inline void gemmGo(hipStream_t s, const GemmParams& p, int Z) {
  dim3 g((p.M + 31) / 32, (p.N + 127) / 128, Z);
  k_gemm<<<g, 128, 0, s>>>(p);
}
static inline void cvt(hipStream_t s, const float* x, h16* y, long long n) {
  long long b = (n + 255) / 256; if (b > 16384) b = 16384;
  k_f32_to_f16<<<(unsigned)b, 256, 0, s>>>(x, y, n);
}

extern "C" void kernel_launch(void* const* d_in, const int* in_sizes, int n_in,
                              void* d_out, int out_size, void* d_ws, size_t ws_size,
                              hipStream_t stream) {
  const float* query     = (const float*)d_in[0];
  const float* key       = (const float*)d_in[1];
  const float* query_pos = (const float*)d_in[3];
  const unsigned char* kpm = (const unsigned char*)d_in[6];
  const float* sa_in_w = (const float*)d_in[8];   const float* sa_in_b = (const float*)d_in[9];
  const float* sa_out_w= (const float*)d_in[10];  const float* sa_out_b= (const float*)d_in[11];
  const float* ln1_g = (const float*)d_in[12];    const float* ln1_b = (const float*)d_in[13];
  const float* ln2_g = (const float*)d_in[14];    const float* ln2_b = (const float*)d_in[15];
  const float* ln3_g = (const float*)d_in[16];    const float* ln3_b = (const float*)d_in[17];
  const float* off_w = (const float*)d_in[18];    const float* off_b = (const float*)d_in[19];
  const float* attw_w= (const float*)d_in[20];    const float* attw_b= (const float*)d_in[21];
  const float* val_w = (const float*)d_in[22];    const float* val_b = (const float*)d_in[23];
  const float* mso_w = (const float*)d_in[24];    const float* mso_b = (const float*)d_in[25];
  const float* ffn_w1= (const float*)d_in[26];    const float* ffn_b1= (const float*)d_in[27];
  const float* ffn_w2= (const float*)d_in[28];    const float* ffn_b2= (const float*)d_in[29];
  const float* norm_g= (const float*)d_in[30];    const float* norm_b= (const float*)d_in[31];
  const float* ref_w1= (const float*)d_in[32];    const float* ref_b1= (const float*)d_in[33];
  const float* ref_w2= (const float*)d_in[34];    const float* ref_b2= (const float*)d_in[35];
  (void)in_sizes; (void)n_in; (void)out_size; (void)ws_size;

  float* out = (float*)d_out;
  const long long OUT2 = (long long)NL_ * MQ * D_;              // 3,686,400
  const long long OUT3 = OUT2 + (long long)BB_ * LEN_ * D_;     // 30,566,400

  // workspace layout
  char* wsb = (char*)d_ws; size_t cur = 0;
  auto take = [&](size_t bytes) -> void* {
    void* p = wsb + cur; cur = (cur + bytes + 255) & ~(size_t)255; return p;
  };
  h16* key16   = (h16*)take((size_t)MKV * D_ * 2);
  h16* qpos16  = (h16*)take((size_t)MQ * D_ * 2);
  h16* w_sain  = (h16*)take((size_t)NL_ * 768 * D_ * 2);
  h16* w_saout = (h16*)take((size_t)NL_ * D_ * D_ * 2);
  h16* w_off   = (h16*)take((size_t)NL_ * 192 * D_ * 2);
  h16* w_attw  = (h16*)take((size_t)NL_ * 96 * D_ * 2);
  h16* w_val   = (h16*)take((size_t)NL_ * D_ * D_ * 2);
  h16* w_mso   = (h16*)take((size_t)NL_ * D_ * D_ * 2);
  h16* w_ffn1  = (h16*)take((size_t)NL_ * DFF_ * D_ * 2);
  h16* w_ffn2  = (h16*)take((size_t)NL_ * D_ * DFF_ * 2);
  h16* w_ref1  = (h16*)take((size_t)D_ * D_ * 2);
  float* tgt   = (float*)take((size_t)MQ * D_ * 4);
  h16* tgt16   = (h16*)take((size_t)MQ * D_ * 2);
  h16* t2_16   = (h16*)take((size_t)MQ * D_ * 2);
  h16* qk16    = (h16*)take((size_t)MQ * D_ * 2);
  h16* QK16    = (h16*)take((size_t)MQ * 512 * 2);
  h16* VT16    = (h16*)take((size_t)BB_ * D_ * KPAD * 2);       // V^T, padded K rows
  float* s32   = (float*)take((size_t)BB_ * NH_ * NQ_ * 304 * 4);
  h16* a16     = (h16*)take((size_t)BB_ * NH_ * NQ_ * KPAD * 2);
  h16* O16     = (h16*)take((size_t)MQ * D_ * 2);
  h16* vdef16  = (h16*)take((size_t)MKV * D_ * 2);
  float* off32 = (float*)take((size_t)MQ * 192 * 4);
  float* atw32 = (float*)take((size_t)MQ * 96 * 4);
  float* aw32  = (float*)take((size_t)MQ * 96 * 4);
  h16* samp16  = (h16*)take((size_t)MQ * D_ * 2);
  h16* h16buf  = (h16*)take((size_t)MQ * DFF_ * 2);
  h16* r116    = (h16*)take((size_t)MQ * D_ * 2);
  float* qrp   = (float*)take((size_t)MQ * 2 * 4);

  // one-time conversions (constant across layers) + VT16 pad zeroing
  cvt(stream, key,       key16,  (long long)MKV * D_);
  cvt(stream, query_pos, qpos16, (long long)MQ * D_);
  cvt(stream, sa_in_w,  w_sain,  (long long)NL_ * 768 * D_);
  cvt(stream, sa_out_w, w_saout, (long long)NL_ * D_ * D_);
  cvt(stream, off_w,    w_off,   (long long)NL_ * 192 * D_);
  cvt(stream, attw_w,   w_attw,  (long long)NL_ * 96 * D_);
  cvt(stream, val_w,    w_val,   (long long)NL_ * D_ * D_);
  cvt(stream, mso_w,    w_mso,   (long long)NL_ * D_ * D_);
  cvt(stream, ffn_w1,   w_ffn1,  (long long)NL_ * DFF_ * D_);
  cvt(stream, ffn_w2,   w_ffn2,  (long long)NL_ * D_ * DFF_);
  cvt(stream, ref_w1,   w_ref1,  (long long)D_ * D_);
  k_zero16<<<4096, 256, 0, stream>>>(VT16, (long long)BB_ * D_ * KPAD);

  hipMemcpyAsync(tgt, query, (size_t)MQ * D_ * 4, hipMemcpyDeviceToDevice, stream);
  hipMemcpyAsync(out + OUT2, key, (size_t)BB_ * LEN_ * D_ * 4,
                 hipMemcpyDeviceToDevice, stream);

  // initial ref points from query_pos
  { GemmParams p = G0();
    p.A = qpos16; p.lda = D_; p.B = w_ref1; p.ldb = D_;
    p.C16 = r116; p.ldc16 = D_; p.bias = ref_b1; p.relu = 1;
    p.M = MQ; p.N = D_; p.K = D_;
    gemmGo(stream, p, 1); }
  k_refhead2<<<(MQ + 127) / 128, 128, 0, stream>>>(r116, ref_w2, ref_b2, qrp, nullptr);

  for (int i = 0; i < NL_; ++i) {
    const h16* wsi  = w_sain  + (size_t)i * 768 * D_;
    const h16* wso  = w_saout + (size_t)i * D_ * D_;
    const h16* wof  = w_off   + (size_t)i * 192 * D_;
    const h16* waw  = w_attw  + (size_t)i * 96 * D_;
    const h16* wvl  = w_val   + (size_t)i * D_ * D_;
    const h16* wms  = w_mso   + (size_t)i * D_ * D_;
    const h16* wf1  = w_ffn1  + (size_t)i * DFF_ * D_;
    const h16* wf2  = w_ffn2  + (size_t)i * D_ * DFF_;

    // ---- self-attention block
    k_layernorm<<<MQ, D_, 0, stream>>>(tgt, ln1_g + i*D_, ln1_b + i*D_,
                                       nullptr, t2_16, query_pos, qk16);
    { GemmParams p = G0();  // Q|K projection (N=512)
      p.A = qk16; p.lda = D_; p.B = wsi; p.ldb = D_;
      p.C16 = QK16; p.ldc16 = 512; p.bias = sa_in_b + i*768;
      p.M = MQ; p.N = 512; p.K = D_;
      gemmGo(stream, p, 1); }
    { GemmParams p = G0();  // V projection -> transposed f16 V^T per batch
      p.A = t2_16; p.lda = D_; p.aZb = (long long)NQ_ * D_;
      p.B = wsi + 512*D_; p.ldb = D_;
      p.CT = VT16; p.ldct = KPAD; p.ctZb = (long long)D_ * KPAD;
      p.bias = sa_in_b + i*768 + 512;
      p.M = NQ_; p.N = D_; p.K = D_;
      gemmGo(stream, p, BB_); }
    { GemmParams p = G0();  // s = Q K^T * DH^-0.5, per (b,h)
      p.A = QK16;       p.lda = 512; p.aZb = (long long)NQ_*512; p.aZh = DH_;
      p.B = QK16 + 256; p.ldb = 512; p.bZb = (long long)NQ_*512; p.bZh = DH_;
      p.C = s32; p.ldc = 304; p.cZb = (long long)NH_*NQ_*304; p.cZh = (long long)NQ_*304;
      p.M = NQ_; p.N = NQ_; p.K = DH_; p.zInner = NH_;
      p.alpha = 0.17677669529663687f;
      gemmGo(stream, p, BB_ * NH_); }
    k_softmax_mha<<<BB_ * NH_ * NQ_, 128, 0, stream>>>(s32, a16);
    { GemmParams p = G0();  // o = a @ V  (B = V^T rows, K padded to 320, no guards)
      p.A = a16; p.lda = KPAD;
      p.aZb = (long long)NH_*NQ_*KPAD; p.aZh = (long long)NQ_*KPAD;
      p.B = VT16; p.ldb = KPAD;
      p.bZb = (long long)D_*KPAD;      p.bZh = (long long)DH_*KPAD;
      p.C16 = O16; p.ldc16 = D_; p.c16Zb = (long long)NQ_*D_; p.c16Zh = DH_;
      p.M = NQ_; p.N = DH_; p.K = KPAD; p.zInner = NH_;
      gemmGo(stream, p, BB_ * NH_); }
    { GemmParams p = G0();  // out projection + residual
      p.A = O16; p.lda = D_; p.B = wso; p.ldb = D_;
      p.C = tgt; p.ldc = D_; p.bias = sa_out_b + i*D_;
      p.res = tgt; p.ldres = D_;
      p.M = MQ; p.N = D_; p.K = D_;
      gemmGo(stream, p, 1); }

    // ---- multi-scale deformable attention block
    k_layernorm<<<MQ, D_, 0, stream>>>(tgt, ln2_g + i*D_, ln2_b + i*D_,
                                       nullptr, t2_16, nullptr, nullptr);
    { GemmParams p = G0();  // value projection of key (masked)
      p.A = key16; p.lda = D_; p.B = wvl; p.ldb = D_;
      p.C16 = vdef16; p.ldc16 = D_; p.bias = val_b + i*D_; p.mask = kpm;
      p.M = MKV; p.N = D_; p.K = D_;
      gemmGo(stream, p, 1); }
    { GemmParams p = G0();  // sampling offsets
      p.A = t2_16; p.lda = D_; p.B = wof; p.ldb = D_;
      p.C = off32; p.ldc = 192; p.bias = off_b + i*192;
      p.M = MQ; p.N = 192; p.K = D_;
      gemmGo(stream, p, 1); }
    { GemmParams p = G0();  // attention weights (pre-softmax)
      p.A = t2_16; p.lda = D_; p.B = waw; p.ldb = D_;
      p.C = atw32; p.ldc = 96; p.bias = attw_b + i*96;
      p.M = MQ; p.N = 96; p.K = D_;
      gemmGo(stream, p, 1); }
    k_attw_softmax<<<(MQ*NH_ + 255)/256, 256, 0, stream>>>(atw32, aw32);
    k_msdeform_sample<<<MQ, 256, 0, stream>>>(vdef16, off32, aw32, qrp, samp16);
    { GemmParams p = G0();  // msdeform output projection + residual
      p.A = samp16; p.lda = D_; p.B = wms; p.ldb = D_;
      p.C = tgt; p.ldc = D_; p.bias = mso_b + i*D_;
      p.res = tgt; p.ldres = D_;
      p.M = MQ; p.N = D_; p.K = D_;
      gemmGo(stream, p, 1); }

    // ---- FFN block
    k_layernorm<<<MQ, D_, 0, stream>>>(tgt, ln3_g + i*D_, ln3_b + i*D_,
                                       nullptr, t2_16, nullptr, nullptr);
    { GemmParams p = G0();  // FFN up + relu
      p.A = t2_16; p.lda = D_; p.B = wf1; p.ldb = D_;
      p.C16 = h16buf; p.ldc16 = DFF_; p.bias = ffn_b1 + i*DFF_; p.relu = 1;
      p.M = MQ; p.N = DFF_; p.K = D_;
      gemmGo(stream, p, 1); }
    { GemmParams p = G0();  // FFN down + residual (also emit tgt f16)
      p.A = h16buf; p.lda = DFF_; p.B = wf2; p.ldb = DFF_;
      p.C = tgt; p.ldc = D_; p.bias = ffn_b2 + i*D_;
      p.res = tgt; p.ldres = D_;
      p.C16 = tgt16; p.ldc16 = D_;
      p.M = MQ; p.N = D_; p.K = DFF_;
      gemmGo(stream, p, 1); }

    // ---- ref head (updates qrp, writes output section 3 for this layer)
    { GemmParams p = G0();
      p.A = tgt16; p.lda = D_; p.B = w_ref1; p.ldb = D_;
      p.C16 = r116; p.ldc16 = D_; p.bias = ref_b1; p.relu = 1;
      p.M = MQ; p.N = D_; p.K = D_;
      gemmGo(stream, p, 1); }
    k_refhead2<<<(MQ + 127) / 128, 128, 0, stream>>>(
        r116, ref_w2, ref_b2, qrp, out + OUT3 + (long long)i * MQ * 6);

    // ---- output section 1: final LayerNorm of tgt
    k_layernorm<<<MQ, D_, 0, stream>>>(tgt, norm_g, norm_b,
                                       out + (long long)i * MQ * D_,
                                       nullptr, nullptr, nullptr);
  }
}
